// hybridSim_20177756357107
// MI455X (gfx1250) — matrix operand order repaired and checked by reading, not compiled, not run on hardware
//
#include <hip/hip_runtime.h>
#include <hip/hip_bf16.h>

// ---------------------------------------------------------------------------
// Hybrid vehicle-dynamics rollout with RNN residual, CDNA5 (gfx1250) WMMA.
//   fc1: (16 traj x 32 feat) @ (32 x 16)  -> one v_wmma_f32_16x16x32_f16
//   per wave: 32 trajectories -> 2 WMMAs per timestep.
// Window kept in LDS as f16 in a 4-slot ring buffer; chronological slot j at
// step t lives in physical slot (t+j)&3, so the shift is free and the WMMA
// A-fragment (lane<16: K=0..7,16..23 ; lane>=16: K=8..15,24..31) is exactly
// two ds_load_b128 per lane.
//
// All LDS sharing is intra-wave (each wave owns its 32 trajectories), and the
// ISA guarantees same-wave LDS ops execute in order, so no s_barrier is
// needed: a compiler-only fence prevents reordering, and waves stay decoupled.
// fc2 weights (wave-uniform) are pinned in SGPRs via readfirstlane.
// ---------------------------------------------------------------------------

typedef __attribute__((ext_vector_type(16))) _Float16 v16h;
typedef __attribute__((ext_vector_type(8)))  _Float16 v8h;
typedef __attribute__((ext_vector_type(8)))  float    v8f;
typedef __attribute__((ext_vector_type(4)))  float    v4f;
typedef __attribute__((ext_vector_type(2)))  float    v2f;

#define BLK 256
#define WIN_STRIDE 40   // halfs per element: 4 slots * 8 + 8 pad (80B, keeps 16B align)
#define Y1_STRIDE 17    // floats per element row, padded to avoid bank conflicts

// Compiler-only fence: same-wave LDS ops are executed in order by hardware,
// we just need to keep the compiler from moving ds ops across this point.
#define WAVE_LDS_FENCE() asm volatile("" ::: "memory")

static __device__ __forceinline__ float fast_tanh(float x) {
#if __has_builtin(__builtin_amdgcn_tanhf)
  return __builtin_amdgcn_tanhf(x);
#elif __has_builtin(__builtin_amdgcn_tanh_f32)
  return __builtin_amdgcn_tanh_f32(x);
#else
  return tanhf(x);
#endif
}

static __device__ __forceinline__ float uniform_load(const float* p) {
  // force wave-uniform value into an SGPR
  return __int_as_float(__builtin_amdgcn_readfirstlane(__float_as_int(*p)));
}

__global__ __launch_bounds__(BLK) void hybrid_rollout_wmma(
    const float* __restrict__ fs,      // (B,4,8) full_states
    const float* __restrict__ act,     // (B,T,2) actions
    const float* __restrict__ fc1w,    // (16,32)
    const float* __restrict__ fc1b,    // (16)
    const float* __restrict__ fc2w,    // (3,16)
    const float* __restrict__ fc2b,    // (3)
    const int*   __restrict__ rnnflag, // scalar
    float*       __restrict__ out,     // (B,T,8)
    int T)
{
  // physics constants (folded at compile time)
  constexpr float Gc  = 9.81f;
  constexpr float Mc  = 0.1667f;
  constexpr float CAF = 5.0f * 0.25f * Mc * Gc;     // == CAR
  constexpr float LF  = 0.09f - 0.036f;
  constexpr float LR  = 0.036f;
  constexpr float IZ  = 0.000267f;
  constexpr float DT  = 0.01f;
  constexpr float THR_OFF = 0.24f, THR_RATIO = 6.98f;
  constexpr float C33 = -(2.f*CAF + 2.f*CAF) / Mc;          // * 1/Vx
  constexpr float C35 =  (2.f*CAF*LF - 2.f*CAF*LR) / Mc;    // a35 = -Vx - C35/Vx
  constexpr float C53 = -(2.f*LF*CAF - 2.f*LR*CAF) / IZ;    // * 1/Vx
  constexpr float C55 = -(2.f*LF*LF*CAF + 2.f*LR*LR*CAF)/IZ;// * 1/Vx
  constexpr float K3  =  2.f*CAF / Mc;                      // * steer
  constexpr float K5  =  2.f*LF*CAF / IZ;                   // * steer
  constexpr float RB0 = 0.05f*Gc, RB1 = 0.05f*Gc, RB2 = 0.05f*Gc*0.2f;
  constexpr float INV_M = 1.0f/Mc, INV_IZ = 1.0f/IZ;
  constexpr float PI = 3.14159265358979323846f;
  constexpr float TWOPI = 2.0f*PI, INV2PI = 1.0f/TWOPI;

  __shared__ __align__(16) _Float16 winLDS[BLK * WIN_STRIDE]; // [elem][slot][8] ring
  __shared__ float y1LDS[BLK * Y1_STRIDE];                    // tanh(fc1) per elem

  const int tid  = threadIdx.x;
  const int lane = tid & 31;
  const int hi   = lane >> 4;        // half-wave id
  const int n    = lane & 15;        // matrix column / row-in-group
  const int wgrp = tid & ~31;        // wave's first element (block-local)
  const long e   = (long)blockIdx.x * BLK + tid; // global trajectory

  const bool rnn = (*rnnflag != 0);

  // --- fc2 weights: wave-uniform, pin to SGPRs once ------------------------
  float w2s[48], b2s[3];
#pragma unroll
  for (int i = 0; i < 48; ++i) w2s[i] = uniform_load(fc2w + i);
#pragma unroll
  for (int i = 0; i < 3; ++i)  b2s[i] = uniform_load(fc2b + i);

  // --- B fragment for fc1 (32x16): column n, K = 8*hi..8*hi+7, 16+8*hi..16+8*hi+7
  v16h bfrag;
  {
    const float* wr = fc1w + n * 32 + 8 * hi;    // fc1_w[n][k] row-major
#pragma unroll
    for (int i = 0; i < 16; ++i) bfrag[i] = (_Float16)wr[i + (i >= 8 ? 8 : 0)];
  }
  // --- C fragment: bias broadcast (C[m][n] = fc1_b[n]) ---------------------
  const float bias = fc1b[n];
  v8f cfrag;
#pragma unroll
  for (int i = 0; i < 8; ++i) cfrag[i] = bias;

  // --- init current state + window -----------------------------------------
  float s0, s1, s2, s3, s4, s5, thr, steer;
  {
    const float* p = fs + e * 32;
#pragma unroll
    for (int j = 0; j < 4; ++j) {        // chron slot j -> phys slot j at t=0
      v8h h;
#pragma unroll
      for (int k = 0; k < 8; ++k) h[k] = (_Float16)p[j * 8 + k];
      *(v8h*)&winLDS[tid * WIN_STRIDE + j * 8] = h;
    }
    s0 = p[24]; s1 = p[25]; s2 = p[26]; s3 = p[27];
    s4 = p[28]; s5 = p[29]; thr = p[30]; steer = p[31];
  }
  WAVE_LDS_FENCE();   // window writes precede first A-fragment reads (same wave)

  for (int t = 0; t < T; ++t) {
    float r0 = 0.f, r1 = 0.f, r2 = 0.f;   // RNN residuals
    if (rnn) {
      // A fragments: lane<16 needs chron slots {0,2}, lane>=16 slots {1,3}
      const int p0 = (t + hi) & 3;
      const int p1 = (t + hi + 2) & 3;
      const int eA0 = wgrp + n;           // rows of WMMA #0
      const int eA1 = eA0 + 16;           // rows of WMMA #1
      v8h a0lo = *(const v8h*)&winLDS[eA0 * WIN_STRIDE + p0 * 8];
      v8h a0hi = *(const v8h*)&winLDS[eA0 * WIN_STRIDE + p1 * 8];
      v8h a1lo = *(const v8h*)&winLDS[eA1 * WIN_STRIDE + p0 * 8];
      v8h a1hi = *(const v8h*)&winLDS[eA1 * WIN_STRIDE + p1 * 8];
      v16h A0 = __builtin_shufflevector(a0lo, a0hi, 0,1,2,3,4,5,6,7,8,9,10,11,12,13,14,15);
      v16h A1 = __builtin_shufflevector(a1lo, a1hi, 0,1,2,3,4,5,6,7,8,9,10,11,12,13,14,15);

      v8f d0 = __builtin_amdgcn_wmma_f32_16x16x32_f16(false, A0, false, bfrag,
                                                      (short)0, cfrag, false, false);
      v8f d1 = __builtin_amdgcn_wmma_f32_16x16x32_f16(false, A1, false, bfrag,
                                                      (short)0, cfrag, false, false);
      // D layout: VGPR j -> row M=j+8*hi, col n. Stage tanh(y1) row-major.
#pragma unroll
      for (int j = 0; j < 8; ++j) {
        y1LDS[(wgrp +      j + 8 * hi) * Y1_STRIDE + n] = fast_tanh(d0[j]);
        y1LDS[(wgrp + 16 + j + 8 * hi) * Y1_STRIDE + n] = fast_tanh(d1[j]);
      }
      WAVE_LDS_FENCE();  // y1 stores precede own-row reads (same wave, in order)

      // fc2: each lane reduces its own trajectory's 16 activations
      float a0 = b2s[0], a1 = b2s[1], a2 = b2s[2];
      const float* y = &y1LDS[tid * Y1_STRIDE];
#pragma unroll
      for (int k = 0; k < 16; ++k) {
        const float yv = y[k];
        a0 += yv * w2s[k];
        a1 += yv * w2s[16 + k];
        a2 += yv * w2s[32 + k];
      }
      r0 = fast_tanh(a0) * RB0;
      r1 = fast_tanh(a1) * RB1;
      r2 = fast_tanh(a2) * RB2;
    }

    // --- bicycle-model dynamics (per-lane f32 VALU) ------------------------
    const float cpsi = __cosf(s4), spsi = __sinf(s4);
    const float Vx = s1 * cpsi + s3 * spsi;          // body-frame v1
    const float v3 = -spsi * s1 + cpsi * s3;
    const float v5 = s5;
    const float invVx = 1.0f / Vx;
    const float d1v = (thr - THR_OFF) * THR_RATIO + r0 * INV_M;
    const float d3v = (C33 * invVx) * v3 + (-Vx - C35 * invVx) * v5
                      + K3 * steer + r1 * INV_M;
    const float d5v = (C53 * invVx) * v3 + (C55 * invVx) * v5
                      + K5 * steer + r2 * INV_IZ;

    const float ns0 = s0 + (cpsi * Vx  - spsi * v3 ) * DT;
    const float ns1 = s1 + (cpsi * d1v - spsi * d3v) * DT;
    const float ns2 = s2 + (spsi * Vx  + cpsi * v3 ) * DT;
    const float ns3 = s3 + (spsi * d1v + cpsi * d3v) * DT;
    const float tp  = s4 + v5 * DT + PI;
    const float ns4 = tp - floorf(tp * INV2PI) * TWOPI - PI;
    const float ns5 = s5 + d5v * DT;

    // actions: read-once stream -> non-temporal
    const v2f a = __builtin_nontemporal_load((const v2f*)&act[(e * T + t) * 2]);

    // --- emit output (B,T,8): write-once stream -> non-temporal ------------
    v4f* op = (v4f*)&out[(e * T + t) * 8];
    __builtin_nontemporal_store((v4f){ns0, ns1, ns2, ns3}, op);
    __builtin_nontemporal_store((v4f){ns4, ns5, a.x, a.y}, op + 1);

    // --- append new state to window ring (overwrites oldest = phys t&3) ---
    {
      v8h h;
      h[0] = (_Float16)ns0; h[1] = (_Float16)ns1;
      h[2] = (_Float16)ns2; h[3] = (_Float16)ns3;
      h[4] = (_Float16)ns4; h[5] = (_Float16)ns5;
      h[6] = (_Float16)a.x; h[7] = (_Float16)a.y;
      *(v8h*)&winLDS[tid * WIN_STRIDE + (t & 3) * 8] = h;
    }
    s0 = ns0; s1 = ns1; s2 = ns2; s3 = ns3; s4 = ns4; s5 = ns5;
    thr = a.x; steer = a.y;
    WAVE_LDS_FENCE();  // window write precedes next step's A-fragment reads
  }
}

extern "C" void kernel_launch(void* const* d_in, const int* in_sizes, int n_in,
                              void* d_out, int out_size, void* d_ws, size_t ws_size,
                              hipStream_t stream) {
  const float* fs   = (const float*)d_in[0];
  const float* act  = (const float*)d_in[1];
  const float* fc1w = (const float*)d_in[2];
  const float* fc1b = (const float*)d_in[3];
  const float* fc2w = (const float*)d_in[4];
  const float* fc2b = (const float*)d_in[5];
  const int*   rnn  = (const int*)d_in[6];
  float* out = (float*)d_out;

  const int B = in_sizes[0] / 32;            // full_states (B,4,8)
  const int T = in_sizes[1] / (2 * B);       // actions (B,T,2)

  dim3 grid((B + BLK - 1) / BLK);            // B=131072 -> 512 blocks, exact
  hybrid_rollout_wmma<<<grid, BLK, 0, stream>>>(fs, act, fc1w, fc1b,
                                               fc2w, fc2b, rnn, out, T);
}
